// SignalDependantNS_30975304139184
// MI455X (gfx1250) — matrix operand order repaired and checked by reading, not compiled, not run on hardware
//
#include <hip/hip_runtime.h>
#include <hip/hip_bf16.h>

typedef _Float16 v16h __attribute__((ext_vector_type(16)));
typedef _Float16 v8h  __attribute__((ext_vector_type(8)));
typedef float    v8f  __attribute__((ext_vector_type(8)));

#define NBINS 10
#define HID   64
#define PCH   116      // C*(3*NUM_BINS-1) = 4*29
#define PCHP  128      // padded to 8 N-tiles of 16
#define HH    256
#define WW    256
#define CC    4
#define BB    16
#define PST   120      // padded p-row stride in LDS floats

// ---------------------------------------------------------------------------
// Repack W2 [116][64][3][3] fp32 -> Bw f16 [tap 9][o 128][ci 64] (zero-padded o)
// ---------------------------------------------------------------------------
__global__ void prep_w2_kernel(const float* __restrict__ W2,
                               _Float16* __restrict__ Bw) {
    int i = blockIdx.x * 256 + threadIdx.x;
    const int total = 9 * PCHP * HID;
    if (i >= total) return;
    int ci = i & 63;
    int r  = i >> 6;
    int o  = r & 127;
    int t  = r >> 7;            // 0..8 = ky*3+kx
    float v = 0.f;
    if (o < PCH) v = W2[(o * HID + ci) * 9 + t];
    Bw[i] = (_Float16)v;        // Bw[(t*128+o)*64+ci] == Bw[i]
}

__global__ void zero_lad_kernel(float* __restrict__ lad) {
    if (threadIdx.x < BB) lad[threadIdx.x] = 0.f;
}

// ---------------------------------------------------------------------------
// Fused: conv1(+ReLU) -> LDS, conv2 via WMMA implicit GEMM, RQ-spline, lad
// One block = 8x8 spatial tile of one batch image. 128 threads = 4 waves.
// ---------------------------------------------------------------------------
__launch_bounds__(128)
__global__ void fused_spline_kernel(const float* __restrict__ x,
                                    const float* __restrict__ clean,
                                    const float* __restrict__ W1,
                                    const float* __restrict__ b1,
                                    const float* __restrict__ b2,
                                    const _Float16* __restrict__ Bw,
                                    float* __restrict__ zout,
                                    float* __restrict__ lad_out) {
    __shared__ float w1s[HID * 36];
    __shared__ float b1s[HID];
    __shared__ float b2s[PCH];
    __shared__ __align__(16) _Float16 hids[10 * 10 * HID]; // 12.8 KB halo tile
    __shared__ __align__(16) float ps[64 * PST];           // p tile 64 px x 116
    __shared__ float red[128];

    const int tid = threadIdx.x;
    const int blk = blockIdx.x;
    const int b  = blk >> 10;
    const int by = (blk >> 5) & 31;
    const int bx = blk & 31;
    const int y0 = by * 8;
    const int x0 = bx * 8;

    for (int i = tid; i < HID * 36; i += 128) w1s[i] = W1[i];
    if (tid < HID) b1s[tid] = b1[tid];
    if (tid < PCH) b2s[tid] = b2[tid];
    __syncthreads();

    // ---- Phase 1: conv1 + ReLU into 10x10x64 f16 LDS (zero outside image) --
    for (int i = tid; i < 100 * HID; i += 128) {
        int co  = i & 63;
        int pix = i >> 6;
        int yy = pix / 10, xx = pix % 10;
        int gy = y0 - 1 + yy;
        int gx = x0 - 1 + xx;
        float acc = 0.f;
        if (gy >= 0 && gy < HH && gx >= 0 && gx < WW) {
            acc = b1s[co];
            #pragma unroll
            for (int ci = 0; ci < CC; ++ci) {
                const float* cp = clean + ((b * CC + ci) * HH) * WW;
                #pragma unroll
                for (int ky = 0; ky < 3; ++ky) {
                    int sy = gy + ky - 1;
                    #pragma unroll
                    for (int kx = 0; kx < 3; ++kx) {
                        int sx = gx + kx - 1;
                        float cv = 0.f;
                        if (sy >= 0 && sy < HH && sx >= 0 && sx < WW)
                            cv = cp[sy * WW + sx];
                        acc += cv * w1s[co * 36 + ci * 9 + ky * 3 + kx];
                    }
                }
            }
            acc = fmaxf(acc, 0.f);
        }
        hids[pix * HID + co] = (_Float16)acc;
    }
    __syncthreads();

    // ---- Phase 2: conv2 as implicit GEMM with v_wmma_f32_16x16x32_f16 ------
    // Wave w owns pixels w*16..w*16+15; accumulates 16 x 128 of p.
    // All 8 B fragments of a (tap, kk) step are loaded as a batch so the
    // scheduler can clause the 16 global_load_b128 and overlap them with the
    // 8 WMMAs (staggered s_wait_loadcnt instead of load->wait0->wmma chains).
    const int wv = tid >> 5;
    const int ln = tid & 31;
    const int lm = ln & 15;      // A: row m ; B: col n
    const int hi = ln >> 4;

    const int pixA = wv * 16 + lm;
    const int ayy = pixA >> 3;
    const int axx = pixA & 7;

    union Frag { v16h v; v8h h[2]; };

    v8f acc[8];
    #pragma unroll
    for (int nt = 0; nt < 8; ++nt)
        #pragma unroll
        for (int e = 0; e < 8; ++e) acc[nt][e] = 0.f;

    // per-lane B base: column o-part (lm) and hi K-chunk folded in once
    const _Float16* blane = Bw + lm * HID + hi * 8;

    for (int t = 0; t < 9; ++t) {
        int dy = t / 3 - 1;
        int dx = t % 3 - 1;
        const _Float16* arow = hids + ((ayy + dy + 1) * 10 + (axx + dx + 1)) * HID;
        const _Float16* btap = blane + (t * PCHP) * HID;
        #pragma unroll
        for (int kk = 0; kk < 2; ++kk) {
            // 16-bit A 16x32 layout: lane(m,hi) holds K = hi*8+(0..7), 16+hi*8+(0..7)
            Frag af;
            af.h[0] = *(const v8h*)(arow + kk * 32 + hi * 8);
            af.h[1] = *(const v8h*)(arow + kk * 32 + 16 + hi * 8);
            // 16-bit B 32x16 layout: lane(n,hi) holds K = hi*8+(0..7), 16+hi*8+(0..7)
            Frag bf[8];
            #pragma unroll
            for (int nt = 0; nt < 8; ++nt) {
                const _Float16* bcol = btap + (nt * 16) * HID + kk * 32;
                bf[nt].h[0] = *(const v8h*)(bcol);
                bf[nt].h[1] = *(const v8h*)(bcol + 16);
            }
            #pragma unroll
            for (int nt = 0; nt < 8; ++nt) {
                acc[nt] = __builtin_amdgcn_wmma_f32_16x16x32_f16(
                    false, af.v, false, bf[nt].v, (short)0, acc[nt], false, false);
            }
        }
    }

    // ---- Phase 3: D tile -> LDS (D layout: lane n = L%16, row = r + 8*hi) --
    #pragma unroll
    for (int nt = 0; nt < 8; ++nt) {
        int o = nt * 16 + lm;
        if (o < PCH) {
            float bias = b2s[o];
            #pragma unroll
            for (int r = 0; r < 8; ++r) {
                int pl = wv * 16 + r + 8 * hi;
                ps[pl * PST + o] = acc[nt][r] + bias;
            }
        }
    }
    __syncthreads();

    // ---- Phase 4: rational-quadratic spline, 256 tasks (64 px x 4 ch) -----
    float ladsum = 0.f;
    #pragma unroll
    for (int s = 0; s < 2; ++s) {
        int task = s * 128 + tid;
        int c  = task & 3;
        int pl = task >> 2;
        int yy = pl >> 3, xx = pl & 7;
        int gidx = ((b * CC + c) * HH + (y0 + yy)) * WW + (x0 + xx);
        float xv = x[gidx];
        const float* pp = ps + pl * PST + c * 29;

        float uw[10], uh[10], dv[11];
        #pragma unroll
        for (int j = 0; j < 10; ++j) uw[j] = pp[j] * 0.125f;       // 1/sqrt(64)
        #pragma unroll
        for (int j = 0; j < 10; ++j) uh[j] = pp[10 + j] * 0.125f;
        dv[0] = 1.0f; dv[10] = 1.0f;   // 0.001 + softplus(log(e^0.999 - 1))
        #pragma unroll
        for (int j = 0; j < 9; ++j) {
            float v = pp[20 + j];
            float sp = (v > 0.f) ? (v + log1pf(expf(-v))) : log1pf(expf(v));
            dv[j + 1] = 0.001f + sp;
        }

        // softmax widths -> knots cw[0..10]
        float mw = uw[0];
        #pragma unroll
        for (int j = 1; j < 10; ++j) mw = fmaxf(mw, uw[j]);
        float ew[10]; float se = 0.f;
        #pragma unroll
        for (int j = 0; j < 10; ++j) { ew[j] = expf(uw[j] - mw); se += ew[j]; }
        float invw = 1.f / se;
        float cw[11]; cw[0] = -1.f;
        float runw = 0.f;
        #pragma unroll
        for (int j = 0; j < 10; ++j) {
            runw += 0.001f + 0.99f * ew[j] * invw;
            cw[j + 1] = -1.f + 2.f * runw;
        }
        cw[10] = 1.f;

        // softmax heights -> knots chh[0..10]
        float mh = uh[0];
        #pragma unroll
        for (int j = 1; j < 10; ++j) mh = fmaxf(mh, uh[j]);
        float eh[10]; float sh = 0.f;
        #pragma unroll
        for (int j = 0; j < 10; ++j) { eh[j] = expf(uh[j] - mh); sh += eh[j]; }
        float invh = 1.f / sh;
        float chh[11]; chh[0] = -1.f;
        float runh = 0.f;
        #pragma unroll
        for (int j = 0; j < 10; ++j) {
            runh += 0.001f + 0.99f * eh[j] * invh;
            chh[j + 1] = -1.f + 2.f * runh;
        }
        chh[10] = 1.f;

        float xin = fminf(fmaxf(xv, -1.f), 1.f);
        int bidx = -1;
        #pragma unroll
        for (int j = 0; j < 11; ++j) bidx += (xin >= cw[j]) ? 1 : 0;
        bidx = min(max(bidx, 0), 9);

        float icw = cw[0], icw1 = cw[1], ich = chh[0], ich1 = chh[1];
        float d0 = dv[0], d1 = dv[1];
        #pragma unroll
        for (int j = 1; j < 10; ++j) {
            bool m = (bidx == j);
            icw  = m ? cw[j]      : icw;
            icw1 = m ? cw[j + 1]  : icw1;
            ich  = m ? chh[j]     : ich;
            ich1 = m ? chh[j + 1] : ich1;
            d0   = m ? dv[j]      : d0;
            d1   = m ? dv[j + 1]  : d1;
        }
        float ibw = icw1 - icw;
        float ihh = ich1 - ich;
        float idl = ihh / ibw;
        float theta = (xin - icw) / ibw;
        float omt = 1.f - theta;
        float tt  = theta * omt;
        float th2 = theta * theta;
        float numer = ihh * (idl * th2 + d0 * tt);
        float denom = idl + (d0 + d1 - 2.f * idl) * tt;
        float zin   = ich + numer / denom;
        float dnum  = idl * idl * (d1 * th2 + 2.f * idl * tt + d0 * omt * omt);
        float ladin = logf(dnum) - 2.f * logf(denom);
        bool inside = (xv >= -1.f) && (xv <= 1.f);
        zout[gidx] = inside ? zin : xv;
        ladsum += inside ? ladin : 0.f;
    }

    // ---- Phase 5: workgroup lad reduction + one atomic per block ----------
    red[tid] = ladsum;
    __syncthreads();
    for (int s2 = 64; s2 > 0; s2 >>= 1) {
        if (tid < s2) red[tid] += red[tid + s2];
        __syncthreads();
    }
    if (tid == 0) atomicAdd(lad_out + b, red[0]);
}

// ---------------------------------------------------------------------------
extern "C" void kernel_launch(void* const* d_in, const int* in_sizes, int n_in,
                              void* d_out, int out_size, void* d_ws, size_t ws_size,
                              hipStream_t stream) {
    (void)in_sizes; (void)n_in; (void)out_size; (void)ws_size;
    const float* x     = (const float*)d_in[0];
    const float* clean = (const float*)d_in[1];
    const float* W1    = (const float*)d_in[2];
    const float* b1    = (const float*)d_in[3];
    const float* W2    = (const float*)d_in[4];
    const float* b2    = (const float*)d_in[5];

    float* z   = (float*)d_out;
    float* lad = z + (size_t)BB * CC * HH * WW;   // 16 trailing floats
    _Float16* Bw = (_Float16*)d_ws;               // 9*128*64 f16 = 147456 B

    prep_w2_kernel<<<(9 * PCHP * HID + 255) / 256, 256, 0, stream>>>(W2, Bw);
    zero_lad_kernel<<<1, 32, 0, stream>>>(lad);
    fused_spline_kernel<<<BB * 32 * 32, 128, 0, stream>>>(x, clean, W1, b1, b2, Bw, z, lad);
}
